// RegDGCNN_88579405513429
// MI455X (gfx1250) — matrix operand order repaired and checked by reading, not compiled, not run on hardware
//
#include <hip/hip_runtime.h>
#include <hip/hip_bf16.h>
#include <math.h>
#include <stdint.h>

typedef __bf16 bf16;
typedef __attribute__((ext_vector_type(16))) __bf16 v16bf;
typedef __attribute__((ext_vector_type(8)))  __bf16 v8bf;
typedef __attribute__((ext_vector_type(8)))  float   v8f;

#define BATCH 4
#define NPTS  2048
#define NTOT  8192            // BATCH*NPTS
#define KSEL  10
#define NHEAD 4
#define EROWS (NTOT*KSEL)     // 81920
#define EMBW  3840
#define EPSV  1e-5f

__device__ inline float bf2f(bf16 h){ return (float)h; }
__device__ inline bf16  f2bf(float f){ return (bf16)f; }

__device__ __forceinline__ unsigned ldsOff(const void* p){
    return (unsigned)(uintptr_t)p;      // low 32 bits of generic LDS addr = LDS byte offset
}
// CDNA5 async copy: global -> LDS, no VGPR round trip, tracked by ASYNCcnt.
__device__ __forceinline__ void asyncLd128(unsigned lds, const void* g){
    unsigned long long ga = (unsigned long long)(uintptr_t)g;
    asm volatile("global_load_async_to_lds_b128 %0, %1, off"
                 :: "v"(lds), "v"(ga) : "memory");
}

// ---------------------------------------------------------------------------
// Core WMMA GEMM:  C[M,Nc] = A[M,Kp] * Bt[Nc,Kp]^T   (bf16 in, f32 accum)
// 128x128 C-tile per 256-thread block (8 waves, each 32x64), BK=32,
// double-buffered LDS filled by global_load_async_to_lds_b128.
// ---------------------------------------------------------------------------
__global__ __launch_bounds__(256)
void gemm_bf16_kernel(const bf16* __restrict__ A, const bf16* __restrict__ Bt,
                      float* __restrict__ Cf, bf16* __restrict__ Cb,
                      int M, int Nc, int Kp,
                      long long sA, long long sB, long long sC)
{
    __shared__ alignas(16) bf16 lA[2][128*32];
    __shared__ alignas(16) bf16 lB[2][128*32];

    const int z = blockIdx.z;
    A  += (long long)z * sA;
    Bt += (long long)z * sB;

    const int tid  = threadIdx.x;
    const int w    = tid >> 5;
    const int lane = tid & 31;
    const int wm   = w & 3;        // 0..3 -> 32-row band
    const int wn   = w >> 2;       // 0..1 -> 64-col band
    const int rowBase = blockIdx.y * 128;
    const int colBase = blockIdx.x * 128;
    const int ml = lane & 15;
    const int hi = lane >> 4;

    // staging addresses: thread covers rows rS and rS+64 at byte-slot (tid&3)*8
    const int rS = tid >> 2, oS = (tid & 3) * 8;
    const bf16* gA0 = A  + (size_t)(rowBase + rS)      * Kp + oS;
    const bf16* gA1 = A  + (size_t)(rowBase + rS + 64) * Kp + oS;
    const bf16* gB0 = Bt + (size_t)(colBase + rS)      * Kp + oS;
    const bf16* gB1 = Bt + (size_t)(colBase + rS + 64) * Kp + oS;
    unsigned dA0[2], dA1[2], dB0[2], dB1[2];
    #pragma unroll
    for (int b = 0; b < 2; ++b) {
        dA0[b] = ldsOff(&lA[b][ rS      *32 + oS]);
        dA1[b] = ldsOff(&lA[b][(rS + 64)*32 + oS]);
        dB0[b] = ldsOff(&lB[b][ rS      *32 + oS]);
        dB1[b] = ldsOff(&lB[b][(rS + 64)*32 + oS]);
    }

    v8f acc[2][4] = {};
    const int nT = Kp >> 5;

    // prologue: stage tile 0 into buffer 0
    asyncLd128(dA0[0], gA0); asyncLd128(dA1[0], gA1);
    asyncLd128(dB0[0], gB0); asyncLd128(dB1[0], gB1);

    for (int t = 0; t < nT; ++t) {
        const int cur = t & 1;
        if (t + 1 < nT) {
            size_t ko = (size_t)(t + 1) * 32;
            asyncLd128(dA0[cur^1], gA0 + ko); asyncLd128(dA1[cur^1], gA1 + ko);
            asyncLd128(dB0[cur^1], gB0 + ko); asyncLd128(dB1[cur^1], gB1 + ko);
            // 4 new in flight; <=4 means tile t's 4 loads have completed
            asm volatile("s_wait_asynccnt 0x4" ::: "memory");
        } else {
            asm volatile("s_wait_asynccnt 0x0" ::: "memory");
        }
        __syncthreads();

        // ---- fragments per ISA 7.12.2 layouts ----
        v16bf aT[2], bT[4];
        #pragma unroll
        for (int ti = 0; ti < 2; ++ti) {
            int m = wm*32 + ti*16 + ml;
            v8bf lo = *(const v8bf*)&lA[cur][m*32 + hi*8];
            v8bf hf = *(const v8bf*)&lA[cur][m*32 + 16 + hi*8];
            v16bf f;
            #pragma unroll
            for (int i = 0; i < 8; ++i) { f[i] = lo[i]; f[8+i] = hf[i]; }
            aT[ti] = f;
        }
        #pragma unroll
        for (int tj = 0; tj < 4; ++tj) {
            int n = wn*64 + tj*16 + ml;
            v8bf p0 = *(const v8bf*)&lB[cur][n*32 + hi*8];
            v8bf p1 = *(const v8bf*)&lB[cur][n*32 + 16 + hi*8];
            v16bf f;
            #pragma unroll
            for (int i = 0; i < 8; ++i) { f[i] = p0[i]; f[8+i] = p1[i]; }
            bT[tj] = f;
        }

        #pragma unroll
        for (int ti = 0; ti < 2; ++ti)
            #pragma unroll
            for (int tj = 0; tj < 4; ++tj)
                acc[ti][tj] = __builtin_amdgcn_wmma_f32_16x16x32_bf16(
                    false, aT[ti], false, bT[tj], (short)0, acc[ti][tj], false, false);

        __syncthreads();
    }

    // ---- epilogue: C layout lanes0-15 M=r, lanes16-31 M=8+r ----
    #pragma unroll
    for (int ti = 0; ti < 2; ++ti)
        #pragma unroll
        for (int tj = 0; tj < 4; ++tj) {
            int gm0 = rowBase + wm*32 + ti*16 + hi*8;
            int gc  = colBase + wn*64 + tj*16 + ml;
            #pragma unroll
            for (int r = 0; r < 8; ++r) {
                size_t off = (size_t)(gm0 + r) * Nc + gc;
                float v = acc[ti][tj][r];
                if (Cb) Cb[(size_t)z*sC + off] = f2bf(v);
                else    Cf[(size_t)z*sC + off] = v;
            }
        }
}

// ---------------------------------------------------------------------------
__global__ void zero_kernel(float* p, int n){
    int t = blockIdx.x*blockDim.x + threadIdx.x;
    if (t < n) p[t] = 0.f;
}

// f32 (rows x cin, stride cin) -> bf16 (rows x Kpad), zero padded
__global__ void cvt_pad_kernel(const float* __restrict__ x, bf16* __restrict__ y,
                               int rows, int cin, int Kpad){
    size_t t = (size_t)blockIdx.x*blockDim.x + threadIdx.x;
    size_t tot = (size_t)rows * Kpad;
    if (t >= tot) return;
    int c = (int)(t % Kpad);
    size_t r = t / Kpad;
    y[t] = (c < cin) ? f2bf(x[r*cin + c]) : f2bf(0.f);
}

// W (fi x fo) f32  ->  Wt (fo x Kpad) bf16, zero padded along K
__global__ void wprep_kernel(const float* __restrict__ W, bf16* __restrict__ Wt,
                             int fi, int fo, int Kpad){
    size_t t = (size_t)blockIdx.x*blockDim.x + threadIdx.x;
    size_t tot = (size_t)fo * Kpad;
    if (t >= tot) return;
    int i = (int)(t % Kpad);
    int o = (int)(t / Kpad);
    Wt[t] = (i < fi) ? f2bf(W[(size_t)i*fo + o]) : f2bf(0.f);
}

__global__ void sqnorm_kernel(const float* __restrict__ x, float* __restrict__ sq,
                              int rows, int cin){
    int i = blockIdx.x*blockDim.x + threadIdx.x;
    if (i >= rows) return;
    float s = 0.f;
    for (int c = 0; c < cin; ++c){ float v = x[(size_t)i*cin + c]; s += v*v; }
    sq[i] = s;
}

// top-K smallest distances per row; gram holds x.x^T per batch
__global__ void topk_kernel(const float* __restrict__ gram, const float* __restrict__ sq,
                            int* __restrict__ idx){
    int i = blockIdx.x*blockDim.x + threadIdx.x;
    if (i >= NTOT) return;
    int bb = i / NPTS;
    int il = i - bb*NPTS;
    const float* g = gram + (size_t)bb*NPTS*NPTS + (size_t)il*NPTS;
    const float* sqb = sq + bb*NPTS;
    float sqi = sq[i];
    int   selI[KSEL];
    #pragma unroll
    for (int k = 0; k < KSEL; ++k) {
        float best = 3.0e38f; int bj = 0;
        for (int j = 0; j < NPTS; ++j) {
            float d = sqi + sqb[j] - 2.0f*g[j];
            if (j == il) d += 1.0e10f;
            bool skip = false;
            #pragma unroll
            for (int t = 0; t < KSEL; ++t)
                if (t < k) skip = skip || (selI[t] == j);
            if (!skip && d < best) { best = d; bj = j; }
        }
        selI[k] = bj;
        idx[(size_t)i*KSEL + k] = bb*NPTS + bj;
    }
}

// e[(i,k),:] = [x_i , x_j - x_i] padded to Kpe, bf16
__global__ void edge_build_kernel(const float* __restrict__ x, const int* __restrict__ idx,
                                  bf16* __restrict__ e, int cin, int Kpe){
    size_t t = (size_t)blockIdx.x*blockDim.x + threadIdx.x;
    size_t tot = (size_t)EROWS * Kpe;
    if (t >= tot) return;
    int c = (int)(t % Kpe);
    size_t row = t / Kpe;
    int i = (int)(row / KSEL);
    int k = (int)(row % KSEL);
    float v;
    if (c < cin) v = x[(size_t)i*cin + c];
    else if (c < 2*cin) {
        int cc = c - cin;
        int j = idx[(size_t)i*KSEL + k];
        v = x[(size_t)j*cin + cc] - x[(size_t)i*cin + cc];
    } else v = 0.f;
    e[t] = f2bf(v);
}

__global__ void bn_stats_kernel(const bf16* __restrict__ X, int rows, int h,
                                float* __restrict__ sum, float* __restrict__ sumsq){
    int c = blockIdx.x*blockDim.x + threadIdx.x;
    if (c >= h) return;
    int chunk = (rows + gridDim.y - 1) / gridDim.y;
    int r0 = blockIdx.y * chunk;
    int r1 = r0 + chunk; if (r1 > rows) r1 = rows;
    float s = 0.f, q = 0.f;
    for (int r = r0; r < r1; ++r) {
        float v = bf2f(X[(size_t)r*h + c]);
        s += v; q += v*v;
    }
    atomicAdd(&sum[c], s);
    atomicAdd(&sumsq[c], q);
}

__global__ void bn_apply_relu_kernel(const bf16* __restrict__ X,
                                     const float* __restrict__ sum, const float* __restrict__ sumsq,
                                     int rows, int h, bf16* __restrict__ Y){
    size_t t = (size_t)blockIdx.x*blockDim.x + threadIdx.x;
    size_t tot = (size_t)rows * h;
    if (t >= tot) return;
    int c = (int)(t % h);
    float inv = 1.0f / rows;
    float m = sum[c] * inv;
    float v = sumsq[c] * inv - m*m;
    float y = (bf2f(X[t]) - m) * rsqrtf(v + EPSV);
    Y[t] = f2bf(y > 0.f ? y : 0.f);
}

// max over K edge rows -> x (f32) and x (bf16)
__global__ void maxk_kernel(const bf16* __restrict__ e, float* __restrict__ xf,
                            bf16* __restrict__ xb, int h){
    size_t t = (size_t)blockIdx.x*blockDim.x + threadIdx.x;
    size_t tot = (size_t)NTOT * h;
    if (t >= tot) return;
    int c = (int)(t % h);
    size_t i = t / h;
    float m = -3.0e38f;
    for (int k = 0; k < KSEL; ++k)
        m = fmaxf(m, bf2f(e[((size_t)i*KSEL + k)*h + c]));
    xf[t] = m;
    xb[t] = f2bf(m);
}

__global__ void colmean_kernel(const float* __restrict__ x, float* __restrict__ cond, int h){
    int c = blockIdx.x*blockDim.x + threadIdx.x;
    if (c >= h) return;
    float s = 0.f;
    for (int i = 0; i < NTOT; ++i) s += x[(size_t)i*h + c];
    cond[c] = s / NTOT;
}

__global__ void matvec_kernel(const float* __restrict__ cond, const float* __restrict__ W,
                              const float* __restrict__ b, float* __restrict__ out, int h){
    int c = blockIdx.x*blockDim.x + threadIdx.x;
    if (c >= h) return;
    float s = b[c];
    for (int k = 0; k < h; ++k) s += cond[k] * W[(size_t)k*h + c];
    out[c] = s;
}

__global__ void film_apply_kernel(float* __restrict__ x, const float* __restrict__ g,
                                  const float* __restrict__ bv, int h, bf16* __restrict__ xb){
    size_t t = (size_t)blockIdx.x*blockDim.x + threadIdx.x;
    size_t tot = (size_t)NTOT * h;
    if (t >= tot) return;
    int c = (int)(t % h);
    float v = g[c]*x[t] + bv[c];
    x[t] = v;
    xb[t] = f2bf(v);
}

__global__ void att1_kernel(const float* __restrict__ hf, const float* __restrict__ aS,
                            const float* __restrict__ aD, int h,
                            float* __restrict__ asrc, float* __restrict__ adst){
    int t = blockIdx.x*blockDim.x + threadIdx.x;
    if (t >= NTOT*NHEAD) return;
    int hd = t & (NHEAD-1);
    int i  = t >> 2;
    int oc = h / NHEAD;
    float s = 0.f, d = 0.f;
    for (int j = 0; j < oc; ++j) {
        float v = hf[(size_t)i*h + hd*oc + j];
        s += v * aS[hd*oc + j];
        d += v * aD[hd*oc + j];
    }
    asrc[t] = s; adst[t] = d;
}

__global__ void att2_kernel(const float* __restrict__ hf, const int* __restrict__ idx,
                            const float* __restrict__ asrc, const float* __restrict__ adst,
                            const float* __restrict__ bias, int h, float* __restrict__ out){
    int i  = blockIdx.x;
    int hd = blockIdx.y;
    int oc = h / NHEAD;
    int nbr[KSEL+1];
    #pragma unroll
    for (int k = 0; k < KSEL; ++k) nbr[k] = idx[(size_t)i*KSEL + k];
    nbr[KSEL] = i;
    float ad = adst[i*NHEAD + hd];
    float lg[KSEL+1];
    float mx = -3.0e38f;
    #pragma unroll
    for (int k = 0; k <= KSEL; ++k) {
        float l = asrc[nbr[k]*NHEAD + hd] + ad;
        l = (l >= 0.f) ? l : 0.2f*l;
        lg[k] = l; mx = fmaxf(mx, l);
    }
    float se = 0.f;
    #pragma unroll
    for (int k = 0; k <= KSEL; ++k) { lg[k] = __expf(lg[k] - mx); se += lg[k]; }
    float inv = 1.0f / se;
    for (int d = threadIdx.x; d < oc; d += blockDim.x) {
        float s = 0.f;
        #pragma unroll
        for (int k = 0; k <= KSEL; ++k)
            s += lg[k] * hf[(size_t)nbr[k]*h + hd*oc + d];
        out[(size_t)i*h + hd*oc + d] = s*inv + bias[hd*oc + d];
    }
}

__global__ void pool_kernel(const float* __restrict__ x, int h, float* __restrict__ emb, int off){
    int c = blockIdx.x*blockDim.x + threadIdx.x;
    int b = blockIdx.y;
    if (c >= h) return;
    float s = 0.f, mx = -3.0e38f;
    for (int n = 0; n < NPTS; ++n) {
        float v = x[((size_t)b*NPTS + n)*h + c];
        s += v; mx = fmaxf(mx, v);
    }
    emb[(size_t)b*EMBW + off + c]     = s / NPTS;
    emb[(size_t)b*EMBW + off + h + c] = mx;
}

__global__ void head_gemm_kernel(const float* __restrict__ in, const float* __restrict__ W,
                                 int Kd, int Nc, float* __restrict__ out){
    int c = blockIdx.x*blockDim.x + threadIdx.x;
    if (c >= Nc) return;
    float a0=0.f,a1=0.f,a2=0.f,a3=0.f;
    for (int k = 0; k < Kd; ++k) {
        float w = W[(size_t)k*Nc + c];
        a0 += in[0*Kd+k]*w; a1 += in[1*Kd+k]*w;
        a2 += in[2*Kd+k]*w; a3 += in[3*Kd+k]*w;
    }
    out[0*Nc+c]=a0; out[1*Nc+c]=a1; out[2*Nc+c]=a2; out[3*Nc+c]=a3;
}

__global__ void head_bn_relu_kernel(float* __restrict__ x, int Nc){
    int c = blockIdx.x*blockDim.x + threadIdx.x;
    if (c >= Nc) return;
    float a0=x[c], a1=x[Nc+c], a2=x[2*Nc+c], a3=x[3*Nc+c];
    float m = (a0+a1+a2+a3)*0.25f;
    float v = (a0*a0+a1*a1+a2*a2+a3*a3)*0.25f - m*m;
    float r = rsqrtf(v + EPSV);
    a0=(a0-m)*r; a1=(a1-m)*r; a2=(a2-m)*r; a3=(a3-m)*r;
    x[c]      = a0>0.f?a0:0.f;  x[Nc+c]   = a1>0.f?a1:0.f;
    x[2*Nc+c] = a2>0.f?a2:0.f;  x[3*Nc+c] = a3>0.f?a3:0.f;
}

__global__ void final_kernel(const float* __restrict__ h2, const float* __restrict__ W3,
                             const float* __restrict__ b3, float* __restrict__ out){
    int b = threadIdx.x;
    if (b >= BATCH) return;
    float s = b3[0];
    for (int k = 0; k < 1024; ++k) s += h2[(size_t)b*1024 + k] * W3[k];
    out[b] = 1.5f / (1.0f + __expf(-s));
}

// ---------------------------------------------------------------------------
extern "C" void kernel_launch(void* const* d_in, const int* in_sizes, int n_in,
                              void* d_out, int out_size, void* d_ws, size_t ws_size,
                              hipStream_t stream) {
    (void)in_sizes; (void)n_in; (void)out_size; (void)ws_size;
    const int CH[5] = {3, 128, 256, 512, 1024};
    const int POOLOFF[4] = {0, 256, 768, 1792};

    const float* pos = (const float*)d_in[0];
    auto F = [&](int i){ return (const float*)d_in[i]; };

    // ---- workspace bump allocator ----
    char* base = (char*)d_ws; size_t off = 0;
    auto alloc = [&](size_t bytes)->void* {
        off = (off + 255) & ~(size_t)255;
        void* p = base + off; off += bytes; return p;
    };
    float* bufA  = (float*)alloc((size_t)NTOT*1024*4);
    float* bufB  = (float*)alloc((size_t)NTOT*1024*4);
    bf16*  xbf   = (bf16*) alloc((size_t)NTOT*1024*2);
    bf16*  ebfA  = (bf16*) alloc((size_t)EROWS*1024*2);
    bf16*  ebfB  = (bf16*) alloc((size_t)EROWS*1024*2);
    float* gram  = (float*)alloc((size_t)BATCH*NPTS*NPTS*4);
    float* hfeat = (float*)alloc((size_t)NTOT*1024*4);
    bf16*  wt    = (bf16*) alloc((size_t)1024*1024*2);
    float* sq    = (float*)alloc((size_t)NTOT*4);
    int*   idx   = (int*)  alloc((size_t)NTOT*KSEL*4);
    float* bsum  = (float*)alloc(1024*4);
    float* bsq   = (float*)alloc(1024*4);
    float* cond  = (float*)alloc(1024*4);
    float* gvec  = (float*)alloc(1024*4);
    float* bvec  = (float*)alloc(1024*4);
    float* asrc  = (float*)alloc((size_t)NTOT*NHEAD*4);
    float* adst  = (float*)alloc((size_t)NTOT*NHEAD*4);
    float* emb   = (float*)alloc((size_t)BATCH*EMBW*4);
    float* h1    = (float*)alloc((size_t)BATCH*2048*4);
    float* h2    = (float*)alloc((size_t)BATCH*1024*4);

    auto cdiv = [](size_t a, size_t b){ return (unsigned)((a + b - 1) / b); };
    auto gemm = [&](const bf16* A, const bf16* Bt, float* Cf, bf16* Cb,
                    int M, int Nc, int Kp, long long sA, long long sB, long long sC, int bat){
        dim3 g(Nc/128, M/128, bat);
        gemm_bf16_kernel<<<g, 256, 0, stream>>>(A, Bt, Cf, Cb, M, Nc, Kp, sA, sB, sC);
    };

    const float* xin = pos;
    for (int s = 0; s < 4; ++s) {
        const int cin = CH[s], h = CH[s+1];
        const int KpadX = (cin + 31) & ~31;
        const int Kpe   = (2*cin + 31) & ~31;
        const int pbase = 1 + s*14;
        const float* mlpW[3] = { F(pbase+0), F(pbase+2), F(pbase+4) };
        const float* gW  = F(pbase+6);  const float* gb  = F(pbase+7);
        const float* bW  = F(pbase+8);  const float* bb  = F(pbase+9);
        const float* gatW = F(pbase+10);
        const float* attS = F(pbase+11), *attD = F(pbase+12), *gatB = F(pbase+13);

        // --- kNN over input x ---
        cvt_pad_kernel<<<cdiv((size_t)NTOT*KpadX,256),256,0,stream>>>(xin, xbf, NTOT, cin, KpadX);
        sqnorm_kernel<<<cdiv(NTOT,256),256,0,stream>>>(xin, sq, NTOT, cin);
        gemm(xbf, xbf, gram, nullptr, NPTS, NPTS, KpadX,
             (long long)NPTS*KpadX, (long long)NPTS*KpadX, (long long)NPTS*NPTS, BATCH);
        topk_kernel<<<cdiv(NTOT,256),256,0,stream>>>(gram, sq, idx);

        // --- edge features + 3x (GEMM -> BN -> ReLU) ---
        edge_build_kernel<<<cdiv((size_t)EROWS*Kpe,256),256,0,stream>>>(xin, idx, ebfA, cin, Kpe);
        for (int l = 0; l < 3; ++l) {
            int fi = (l == 0) ? 2*cin : h;
            int Kp = (l == 0) ? Kpe   : h;
            wprep_kernel<<<cdiv((size_t)h*Kp,256),256,0,stream>>>(mlpW[l], wt, fi, h, Kp);
            gemm(ebfA, wt, nullptr, ebfB, EROWS, h, Kp, 0, 0, 0, 1);
            zero_kernel<<<cdiv(h,256),256,0,stream>>>(bsum, h);
            zero_kernel<<<cdiv(h,256),256,0,stream>>>(bsq, h);
            bn_stats_kernel<<<dim3(cdiv(h,256),160),256,0,stream>>>(ebfB, EROWS, h, bsum, bsq);
            bn_apply_relu_kernel<<<cdiv((size_t)EROWS*h,256),256,0,stream>>>(ebfB, bsum, bsq, EROWS, h, ebfA);
        }
        maxk_kernel<<<cdiv((size_t)NTOT*h,256),256,0,stream>>>(ebfA, bufB, xbf, h);

        // --- FiLM ---
        colmean_kernel<<<cdiv(h,256),256,0,stream>>>(bufB, cond, h);
        matvec_kernel<<<cdiv(h,256),256,0,stream>>>(cond, gW, gb, gvec, h);
        matvec_kernel<<<cdiv(h,256),256,0,stream>>>(cond, bW, bb, bvec, h);
        film_apply_kernel<<<cdiv((size_t)NTOT*h,256),256,0,stream>>>(bufB, gvec, bvec, h, xbf);

        // --- GAT ---
        sqnorm_kernel<<<cdiv(NTOT,256),256,0,stream>>>(bufB, sq, NTOT, h);
        gemm(xbf, xbf, gram, nullptr, NPTS, NPTS, h,
             (long long)NPTS*h, (long long)NPTS*h, (long long)NPTS*NPTS, BATCH);
        topk_kernel<<<cdiv(NTOT,256),256,0,stream>>>(gram, sq, idx);
        wprep_kernel<<<cdiv((size_t)h*h,256),256,0,stream>>>(gatW, wt, h, h, h);
        gemm(xbf, wt, hfeat, nullptr, NTOT, h, h, 0, 0, 0, 1);
        att1_kernel<<<cdiv(NTOT*NHEAD,256),256,0,stream>>>(hfeat, attS, attD, h, asrc, adst);
        att2_kernel<<<dim3(NTOT,NHEAD),64,0,stream>>>(hfeat, idx, asrc, adst, gatB, h, bufA);

        // --- global pooling into emb ---
        pool_kernel<<<dim3(cdiv(h,256),BATCH),256,0,stream>>>(bufA, h, emb, POOLOFF[s]);

        xin = bufA;   // next stage input
    }

    // --- head MLP (biases before BN cancel; lin3 bias kept) ---
    const float* lin1W = F(57); const float* lin2W = F(59);
    const float* lin3W = F(61); const float* lin3b = F(62);
    head_gemm_kernel<<<cdiv(2048,256),256,0,stream>>>(emb, lin1W, EMBW, 2048, h1);
    head_bn_relu_kernel<<<cdiv(2048,256),256,0,stream>>>(h1, 2048);
    head_gemm_kernel<<<cdiv(1024,256),256,0,stream>>>(h1, lin2W, 2048, 1024, h2);
    head_bn_relu_kernel<<<cdiv(1024,256),256,0,stream>>>(h2, 1024);
    final_kernel<<<1,64,0,stream>>>(h2, lin3W, lin3b, (float*)d_out);
}
